// PointerNetWork_4535485464619
// MI455X (gfx1250) — matrix operand order repaired and checked by reading, not compiled, not run on hardware
//
#include <hip/hip_runtime.h>
#include <hip/hip_bf16.h>

// ---------------------------------------------------------------------------
// Types for CDNA5 WMMA (wave32): A/B = 16 bf16 (8 VGPRs), C/D = 8 f32.
// ---------------------------------------------------------------------------
typedef __attribute__((ext_vector_type(16))) __bf16 v16bf;
typedef __attribute__((ext_vector_type(8)))  float  v8f;

union AFrag { v16bf v; unsigned short u[16]; uint4 q[2]; };
union BFrag { v16bf v; uint4 q[2]; };

__device__ __forceinline__ unsigned short f2bf(float f) {
    unsigned int x = __float_as_uint(f);
    return (unsigned short)((x + 0x7FFFu + ((x >> 16) & 1u)) >> 16); // RNE
}
__device__ __forceinline__ float bf2f(unsigned short u) {
    return __uint_as_float(((unsigned int)u) << 16);
}

// Scheduling fence: keep next-chunk loads issued before current-chunk WMMAs.
__device__ __forceinline__ void sched_fence() {
#if __has_builtin(__builtin_amdgcn_sched_barrier)
    __builtin_amdgcn_sched_barrier(0);
#endif
}

// Hardware V_TANH_F32 (CDNA5 trans op); builtin if declared, else inline asm.
__device__ __forceinline__ float fast_tanh(float x) {
#if __has_builtin(__builtin_amdgcn_tanhf)
    return __builtin_amdgcn_tanhf(x);
#else
    float r;
    asm("v_tanh_f32 %0, %1" : "=v"(r) : "v"(x));
    return r;
#endif
}
__device__ __forceinline__ float fast_sigmoid(float x) {
    return __builtin_amdgcn_rcpf(1.f + __builtin_amdgcn_exp2f(x * -1.4426950408889634f));
}

#define NEG_FILL (-1.0e9f)
#define PROB_MIN (1e-9f)

// ---------------------------------------------------------------------------
// Weight pre-pack: B matrix [K,N] (B[k,n] = W[n,k], optional concat along k)
// into WMMA-native fragment layout:
//   chunk c = k/32, ntile = n/16, lane = (n%16)+16*((k/8)%2), elem = k%8 + 8*((k%32)>=16)
//   offset  = ((c*(N/16)+ntile)*32 + lane)*16 + elem   (16 bf16/lane contiguous)
// ---------------------------------------------------------------------------
__global__ void pack_b_kernel(const float* __restrict__ src0,   // [N, k0]
                              const float* __restrict__ src1,   // [N, K-k0] or null
                              unsigned short* __restrict__ out,
                              int k0, int K, int N)
{
    int idx = blockIdx.x * blockDim.x + threadIdx.x;
    if (idx >= K * N) return;
    int k = idx / N, n = idx % N;
    float v = (k < k0) ? src0[(long)n * k0 + k]
                       : src1[(long)n * (K - k0) + (k - k0)];
    int c    = k >> 5;
    int kk   = k & 31;
    int lane = (n & 15) + ((kk & 8) ? 16 : 0);
    int elem = (kk & 7) + ((kk & 16) ? 8 : 0);
    long off = (((long)c * (N >> 4) + (n >> 4)) * 32 + lane) * 16 + elem;
    out[off] = f2bf(v);
}

__global__ void f32_to_bf16_kernel(const float* __restrict__ in,
                                   unsigned short* __restrict__ out, long n)
{
    long i = (long)blockIdx.x * blockDim.x + threadIdx.x;
    if (i < n) out[i] = f2bf(in[i]);
}

__global__ void add_bias_kernel(const float* __restrict__ a,
                                const float* __restrict__ b,
                                float* __restrict__ out, int n)
{
    int i = blockIdx.x * blockDim.x + threadIdx.x;
    if (i < n) out[i] = a[i] + b[i];
}

__global__ void init_enc_kernel(unsigned short* __restrict__ h,
                                float* __restrict__ c, int n)
{
    int i = blockIdx.x * blockDim.x + threadIdx.x;
    if (i < n) { h[i] = 0; c[i] = 0.f; }
}

__global__ void init_dec_kernel(const float* __restrict__ h0, const float* __restrict__ c0,
                                unsigned short* __restrict__ h, float* __restrict__ c,
                                unsigned short* __restrict__ dec_in, int* __restrict__ selected,
                                int nBH, int nBD, int nBT)
{
    int i = blockIdx.x * blockDim.x + threadIdx.x;
    if (i < nBH) { h[i] = f2bf(h0[i]); c[i] = c0[i]; }
    if (i < nBD) dec_in[i] = 0;
    if (i < nBT) selected[i] = 0;
}

#define WMMA_BF16(ACC, A, B) \
    (ACC) = __builtin_amdgcn_wmma_f32_16x16x32_bf16(false, (A).v, false, (B).v, (short)0, (ACC), false, false)

// ---------------------------------------------------------------------------
// Fully-unrolled, double-buffered K loop: 4 output tiles per wave.
// NCH / NTILES / K0 are compile-time so after unrolling:
//  - the concat source select (chunk vs K0) folds away (no v_cndmask),
//  - all chunk offsets become 24-bit instruction-immediate offsets off a few
//    base pointers (2 A bases with khalf folded in, 4 B tile bases),
//  - the ping-pong arrays become renamed registers (no v_mov copies).
// A sched_barrier pins next-chunk loads ahead of current-chunk WMMAs so RA
// keeps both buffers live and WMMA waits refer to chunk-old (complete) loads.
// A layout (16-bit, 16x32): lane<16 holds K runs [kb..kb+7],[kb+16..kb+23],
// kb = chunk base (+8 for lanes>=16). Each run = one b128 load.
// ---------------------------------------------------------------------------
template <int NCH, int NTILES, bool CONCAT, int K0>
__device__ __forceinline__ void gemm_k_loop(
    const unsigned short* __restrict__ A0, int lda0,
    const unsigned short* __restrict__ A1, int lda1,
    const unsigned short* __restrict__ Bp,
    int nt0, int nt1, int nt2, int nt3,
    int mrow, int khalf, int lane,
    v8f& acc0, v8f& acc1, v8f& acc2, v8f& acc3)
{
    const unsigned short* baseX = A0 + (long)mrow * lda0 + khalf;
    const unsigned short* baseH = CONCAT ? (A1 + (long)mrow * lda1 + khalf) : nullptr;
    const unsigned short* bb0 = Bp + ((long)nt0 * 32 + lane) * 16;
    const unsigned short* bb1 = Bp + ((long)nt1 * 32 + lane) * 16;
    const unsigned short* bb2 = Bp + ((long)nt2 * 32 + lane) * 16;
    const unsigned short* bb3 = Bp + ((long)nt3 * 32 + lane) * 16;

    auto load_a_c = [&](int cc) -> AFrag {
        const int kchunk = cc << 5;                       // constant after unroll
        const bool inX = (!CONCAT) || (kchunk < K0);      // compile-time select
        const unsigned short* bp = inX ? (baseX + kchunk) : (baseH + (kchunk - K0));
        AFrag a_;
        a_.q[0] = *(const uint4*)(bp);
        a_.q[1] = *(const uint4*)(bp + 16);
        return a_;
    };
    auto load_b_c = [&](const unsigned short* bb, int cc) -> BFrag {
        const long off = (long)cc * NTILES * 512;         // constant after unroll
        BFrag b_;
        b_.q[0] = *(const uint4*)(bb + off);
        b_.q[1] = *(const uint4*)(bb + off + 8);
        return b_;
    };

    AFrag a[2];
    BFrag b[2][4];
    a[0]    = load_a_c(0);
    b[0][0] = load_b_c(bb0, 0);
    b[0][1] = load_b_c(bb1, 0);
    b[0][2] = load_b_c(bb2, 0);
    b[0][3] = load_b_c(bb3, 0);
#pragma unroll
    for (int c = 0; c < NCH; ++c) {
        const int cur = c & 1;
        const int nxt = cur ^ 1;
        if (c + 1 < NCH) {
            a[nxt]    = load_a_c(c + 1);
            b[nxt][0] = load_b_c(bb0, c + 1);
            b[nxt][1] = load_b_c(bb1, c + 1);
            b[nxt][2] = load_b_c(bb2, c + 1);
            b[nxt][3] = load_b_c(bb3, c + 1);
        }
        sched_fence();   // loads for chunk c+1 stay above WMMAs for chunk c
        WMMA_BF16(acc0, a[cur], b[cur][0]);
        WMMA_BF16(acc1, a[cur], b[cur][1]);
        WMMA_BF16(acc2, a[cur], b[cur][2]);
        WMMA_BF16(acc3, a[cur], b[cur][3]);
        sched_fence();   // keep WMMAs from sinking below next iteration's loads
    }
}

// ---------------------------------------------------------------------------
// Fused LSTM step: gates = [x|h_in] @ Wcat + bias, full cell update in-register.
// Wave's 4 accumulators = the 4 gate sections (i,f,g,o) at the SAME hidden
// offset j0, so each lane holds i,f,g,o for one (row, j).
// h ping-pongs in bf16 (read h_in, write h_out); c (f32) updates in place.
// grid = (B/16, H/64), block = 128 (4 waves). K = 128+256 -> 12 chunks.
// ---------------------------------------------------------------------------
__global__ __launch_bounds__(128) void lstm_step_fused(
    const unsigned short* __restrict__ X, int ldx,           // x part bf16 [B, 128]
    const unsigned short* __restrict__ h_in,                 // bf16 [B, H]
    const unsigned short* __restrict__ Bp, int H,
    const float* __restrict__ bias,                          // [4H]
    unsigned short* __restrict__ h_out, float* __restrict__ c_io)
{
    const int wave  = threadIdx.x >> 5;
    const int lane  = threadIdx.x & 31;
    const int m0    = blockIdx.x << 4;
    const int j0    = (blockIdx.y << 6) + (wave << 4);
    const int jt    = j0 >> 4;
    const int mrow  = m0 + (lane & 15);
    const int khalf = (lane >> 4) << 3;
    const int hseg  = H >> 4;

    v8f acc0 = {}, acc1 = {}, acc2 = {}, acc3 = {};
    gemm_k_loop<12, 64, true, 128>(X, ldx, h_in, H, Bp,
                                   0 * hseg + jt, 1 * hseg + jt, 2 * hseg + jt, 3 * hseg + jt,
                                   mrow, khalf, lane, acc0, acc1, acc2, acc3);

    const int rowlo = m0 + ((lane >> 4) << 3);
    const int j     = j0 + (lane & 15);
    const float bi_ = bias[j];
    const float bf_ = bias[H + j];
    const float bg_ = bias[2 * H + j];
    const float bo_ = bias[3 * H + j];
    const long  base = (long)rowlo * H + j;
#pragma unroll
    for (int r = 0; r < 8; ++r) {
        const long off = base + (long)r * H;
        float gi = acc0[r] + bi_;
        float gf = acc1[r] + bf_;
        float gg = acc2[r] + bg_;
        float go = acc3[r] + bo_;
        float cn = fast_sigmoid(gf) * c_io[off] + fast_sigmoid(gi) * fast_tanh(gg);
        c_io[off]  = cn;
        h_out[off] = f2bf(fast_sigmoid(go) * fast_tanh(cn));
    }
}

// ---------------------------------------------------------------------------
// Blend GEMM: out_bf16[M,256] = A_bf16[M,256] @ Bpacked.  K = 256 -> 8 chunks.
// grid = (M/16, 1), block = 128 (4 waves x 64 cols).
// ---------------------------------------------------------------------------
__global__ __launch_bounds__(128) void blend_gemm(
    const unsigned short* __restrict__ A, int lda,
    const unsigned short* __restrict__ Bp,
    unsigned short* __restrict__ out, long ostride)
{
    const int wave  = threadIdx.x >> 5;
    const int lane  = threadIdx.x & 31;
    const int m0    = blockIdx.x << 4;
    const int nwave = wave << 6;
    const int mrow  = m0 + (lane & 15);
    const int khalf = (lane >> 4) << 3;
    const int nt    = nwave >> 4;

    v8f acc0 = {}, acc1 = {}, acc2 = {}, acc3 = {};
    gemm_k_loop<8, 16, false, 0>(A, lda, nullptr, 0, Bp,
                                 nt, nt + 1, nt + 2, nt + 3,
                                 mrow, khalf, lane, acc0, acc1, acc2, acc3);

    const int rowlo = m0 + ((lane >> 4) << 3);
    const int coln  = lane & 15;
#pragma unroll
    for (int jj = 0; jj < 4; ++jj) {
        const v8f acc = (jj == 0) ? acc0 : (jj == 1) ? acc1 : (jj == 2) ? acc2 : acc3;
        const int ncol = nwave + (jj << 4) + coln;
#pragma unroll
        for (int r = 0; r < 8; ++r)
            out[(long)(rowlo + r) * ostride + ncol] = f2bf(acc[r]);
    }
}

// ---------------------------------------------------------------------------
// Decoder attention step (one block per batch row b, 256 threads = 8 waves):
//   scores[t] = vt . tanh(blend1[b,t,:] + blend2[b,:])  (hardware V_TANH_F32)
//   mask selected -> argmax (first occurrence) -> softmax -> clamp
//   write probs[b, step, :], mark selected, gather dec_in = targets_bf16[b,sel,:]
// ---------------------------------------------------------------------------
__global__ __launch_bounds__(256) void attn_step(
    const unsigned short* __restrict__ blend1,   // [B,T,W] bf16
    const unsigned short* __restrict__ blend2,   // [B,W] bf16
    const float* __restrict__ vt,                // [W]
    const unsigned short* __restrict__ tgt_bf16, // [B,T,D] bf16
    int* __restrict__ selected,                  // [B,T]
    unsigned short* __restrict__ dec_in,         // [B,D] bf16
    float* __restrict__ probs,                   // [B,T,T]
    int step, int B, int T, int D, int W)
{
    __shared__ float s_b2[256];
    __shared__ float s_vt[256];
    __shared__ float s_sc[128];
    __shared__ int   s_sel;

    const int b   = blockIdx.x;
    const int tid = threadIdx.x;
    if (tid < W) {
        s_b2[tid] = bf2f(blend2[(long)b * W + tid]);
        s_vt[tid] = vt[tid];
    }
    __syncthreads();

    const int wave = tid >> 5, lane = tid & 31;
    for (int t = wave; t < T; t += 8) {
        const unsigned short* rowp = blend1 + ((long)(b * T + t)) * W + lane * 8;
        if (t + 8 < T)
            __builtin_prefetch(blend1 + ((long)(b * T + t + 8)) * W + lane * 8, 0, 0);
        uint4 q = *(const uint4*)rowp;
        unsigned int uu[4] = { q.x, q.y, q.z, q.w };
        float s = 0.f;
#pragma unroll
        for (int e = 0; e < 4; ++e) {
            int w0 = lane * 8 + e * 2;
            float lo = __uint_as_float(uu[e] << 16);
            float hi = __uint_as_float(uu[e] & 0xFFFF0000u);
            s += fast_tanh(lo + s_b2[w0])     * s_vt[w0];
            s += fast_tanh(hi + s_b2[w0 + 1]) * s_vt[w0 + 1];
        }
        for (int off = 16; off; off >>= 1) s += __shfl_xor(s, off, 32);
        if (lane == 0) s_sc[t] = s;
    }
    __syncthreads();

    if (tid < 32) {
        float vloc[4];
        float best = -3.4e38f; int bidx = 0;
#pragma unroll
        for (int j = 0; j < 4; ++j) {
            int t = tid + 32 * j;
            float sc = -3.4e38f;
            if (t < T) sc = selected[(long)b * T + t] ? NEG_FILL : s_sc[t];
            vloc[j] = sc;
            if (sc > best) { best = sc; bidx = t; }   // ascending t: first occurrence
        }
        float bv = best; int bi = bidx;
        for (int off = 16; off; off >>= 1) {
            float ov = __shfl_xor(bv, off, 32);
            int   oi = __shfl_xor(bi, off, 32);
            if (ov > bv || (ov == bv && oi < bi)) { bv = ov; bi = oi; }
        }
        if (lane == 0) { s_sel = bi; selected[(long)b * T + bi] = 1; }

        float e4[4]; float sum = 0.f;
#pragma unroll
        for (int j = 0; j < 4; ++j) {
            int t = tid + 32 * j;
            float e = (t < T) ? __expf(vloc[j] - bv) : 0.f;
            e4[j] = e; sum += e;
        }
        for (int off = 16; off; off >>= 1) sum += __shfl_xor(sum, off, 32);
        float inv = __builtin_amdgcn_rcpf(sum);
#pragma unroll
        for (int j = 0; j < 4; ++j) {
            int t = tid + 32 * j;
            if (t < T)
                probs[((long)b * T + step) * T + t] = fmaxf(e4[j] * inv, PROB_MIN);
        }
    }
    __syncthreads();

    if (tid < D) dec_in[(long)b * D + tid] = tgt_bf16[((long)b * T + s_sel) * D + tid];
}

// ---------------------------------------------------------------------------
// Host driver (graph-capture safe: only kernel launches on `stream`).
// ---------------------------------------------------------------------------
extern "C" void kernel_launch(void* const* d_in, const int* in_sizes, int n_in,
                              void* d_out, int out_size, void* d_ws, size_t ws_size,
                              hipStream_t stream)
{
    (void)in_sizes; (void)n_in; (void)out_size; (void)ws_size;
    const int B = 2048, T = 100, D = 128, H = 256, Wd = 256;
    const int G = 4 * H;          // 1024 gate width
    const int Kc = D + H;         // 384 concat K (12 chunks of 32)

    const float* targets  = (const float*)d_in[0];
    const float* h0       = (const float*)d_in[1];
    const float* c0       = (const float*)d_in[2];
    const float* enc_w_ih = (const float*)d_in[3];
    const float* enc_w_hh = (const float*)d_in[4];
    const float* enc_b_ih = (const float*)d_in[5];
    const float* enc_b_hh = (const float*)d_in[6];
    const float* dec_w_ih = (const float*)d_in[7];
    const float* dec_w_hh = (const float*)d_in[8];
    const float* dec_b_ih = (const float*)d_in[9];
    const float* dec_b_hh = (const float*)d_in[10];
    const float* W1       = (const float*)d_in[11];
    const float* W2       = (const float*)d_in[12];
    const float* vt       = (const float*)d_in[13];
    float* probs = (float*)d_out;

    char* p = (char*)d_ws;
    auto alloc = [&](size_t bytes) -> void* {
        void* r = (void*)p;
        p += (bytes + 255) & ~(size_t)255;
        return r;
    };
    unsigned short* wcat_enc = (unsigned short*)alloc((size_t)Kc * G * 2);
    unsigned short* wcat_dec = (unsigned short*)alloc((size_t)Kc * G * 2);
    unsigned short* w1p      = (unsigned short*)alloc((size_t)H * Wd * 2);
    unsigned short* w2p      = (unsigned short*)alloc((size_t)H * Wd * 2);
    float* b_enc = (float*)alloc((size_t)G * 4);
    float* b_dec = (float*)alloc((size_t)G * 4);
    unsigned short* tgt_bf16 = (unsigned short*)alloc((size_t)B * T * D * 2); // 52 MB
    unsigned short* hbuf0    = (unsigned short*)alloc((size_t)B * H * 2);
    unsigned short* hbuf1    = (unsigned short*)alloc((size_t)B * H * 2);
    float* c = (float*)alloc((size_t)B * H * 4);
    unsigned short* blend1 = (unsigned short*)alloc((size_t)B * T * Wd * 2);  // 100 MB
    unsigned short* blend2 = (unsigned short*)alloc((size_t)B * Wd * 2);
    unsigned short* dec_in = (unsigned short*)alloc((size_t)B * D * 2);
    int* selected = (int*)alloc((size_t)B * T * 4);

    // ---- one-time prep ----
    const long nTgt = (long)B * T * D;
    f32_to_bf16_kernel<<<(unsigned)((nTgt + 255) / 256), 256, 0, stream>>>(targets, tgt_bf16, nTgt);
    pack_b_kernel<<<(Kc * G + 255) / 256, 256, 0, stream>>>(enc_w_ih, enc_w_hh, wcat_enc, D, Kc, G);
    pack_b_kernel<<<(Kc * G + 255) / 256, 256, 0, stream>>>(dec_w_ih, dec_w_hh, wcat_dec, D, Kc, G);
    pack_b_kernel<<<(H * Wd + 255) / 256, 256, 0, stream>>>(W1, nullptr, w1p, H, H, Wd);
    pack_b_kernel<<<(H * Wd + 255) / 256, 256, 0, stream>>>(W2, nullptr, w2p, H, H, Wd);
    add_bias_kernel<<<(G + 255) / 256, 256, 0, stream>>>(enc_b_ih, enc_b_hh, b_enc, G);
    add_bias_kernel<<<(G + 255) / 256, 256, 0, stream>>>(dec_b_ih, dec_b_hh, b_dec, G);
    init_enc_kernel<<<(B * H + 255) / 256, 256, 0, stream>>>(hbuf0, c, B * H);

    dim3 gLstm(B / 16, H / 64);    // (128, 4): 4 gate tiles per wave at same j
    dim3 gBlend(B / 16, 1);
    int initBlocks = (B * H + 255) / 256;

    unsigned short* ha = hbuf0;
    unsigned short* hb = hbuf1;

    // ---- encoder: fused LSTM step -> blend1 slice GEMM ----
    for (int t = 0; t < T; ++t) {
        lstm_step_fused<<<gLstm, 128, 0, stream>>>(
            tgt_bf16 + (size_t)t * D, T * D, ha, wcat_enc, H, b_enc, hb, c);
        blend_gemm<<<gBlend, 128, 0, stream>>>(
            hb, H, w1p, blend1 + (size_t)t * Wd, (long)T * Wd);
        unsigned short* tmp = ha; ha = hb; hb = tmp;
    }

    // ---- decoder init ----
    init_dec_kernel<<<initBlocks, 256, 0, stream>>>(h0, c0, ha, c, dec_in, selected,
                                                    B * H, B * D, B * T);

    // ---- decoder: fused LSTM step -> blend2 GEMM -> attention ----
    for (int s = 0; s < T; ++s) {
        lstm_step_fused<<<gLstm, 128, 0, stream>>>(
            dec_in, D, ha, wcat_dec, H, b_dec, hb, c);
        blend_gemm<<<gBlend, 128, 0, stream>>>(
            hb, H, w2p, blend2, (long)Wd);
        attn_step<<<B, 256, 0, stream>>>(blend1, blend2, vt, tgt_bf16, selected,
                                         dec_in, probs, s, B, T, D, Wd);
        unsigned short* tmp = ha; ha = hb; hb = tmp;
    }
}
